// Controller_XNOR_67302137528691
// MI455X (gfx1250) — 8-bit matrix operand order repaired and checked by reading, compiled, not run on hardware
//
#include <hip/hip_runtime.h>
#include <hip/hip_bf16.h>
#include <stdint.h>

typedef __attribute__((ext_vector_type(8))) int v8i;

__global__ void pack_sign_i8_kernel(const float* __restrict__ w,
                                    signed char* __restrict__ dst, int n) {
    int i = blockIdx.x * blockDim.x + threadIdx.x;
    if (i < n) {
        float v = w[i];
        dst[i] = (signed char)((v > 0.0f) - (v < 0.0f));
    }
}

__global__ void conv1_relu_kernel(const float* __restrict__ x,
                                  const float* __restrict__ w,
                                  float* __restrict__ out, int total) {
    int idx = blockIdx.x * blockDim.x + threadIdx.x;
    if (idx >= total) return;
    int p = idx % 784;
    int o = (idx / 784) % 128;
    int b = idx / (784 * 128);
    int oy = p / 28, ox = p % 28;
    const float* xp = x + (size_t)b * 1024 + oy * 32 + ox;
    const float* wp = w + o * 25;
    float s = 0.f;
#pragma unroll
    for (int ky = 0; ky < 5; ++ky)
#pragma unroll
        for (int kx = 0; kx < 5; ++kx)
            s = fmaf(xp[ky * 32 + kx], wp[ky * 5 + kx], s);
    out[idx] = s > 0.f ? s : 0.f;
}

__global__ void reduce_sum_partial_kernel(const float* __restrict__ src,
                                          float* __restrict__ partial, int Nelem) {
    __shared__ float sm[256];
    int b = blockIdx.y;
    const float* s = src + (size_t)b * Nelem;
    float acc = 0.f;
    for (int i = blockIdx.x * blockDim.x + threadIdx.x; i < Nelem;
         i += gridDim.x * blockDim.x)
        acc += s[i];
    sm[threadIdx.x] = acc;
    __syncthreads();
    for (int st = 128; st > 0; st >>= 1) {
        if ((int)threadIdx.x < st) sm[threadIdx.x] += sm[threadIdx.x + st];
        __syncthreads();
    }
    if (threadIdx.x == 0) partial[(size_t)b * gridDim.x + blockIdx.x] = sm[0];
}

__global__ void reduce_sum_final_kernel(const float* __restrict__ partial,
                                        float* __restrict__ stats, int nchunks) {
    __shared__ float sm[256];
    int b = blockIdx.x;
    float acc = 0.f;
    for (int i = threadIdx.x; i < nchunks; i += blockDim.x)
        acc += partial[(size_t)b * nchunks + i];
    sm[threadIdx.x] = acc;
    __syncthreads();
    for (int st = 128; st > 0; st >>= 1) {
        if ((int)threadIdx.x < st) sm[threadIdx.x] += sm[threadIdx.x + st];
        __syncthreads();
    }
    if (threadIdx.x == 0) stats[b] = sm[0];
}

__global__ void binarize_kernel(const float* __restrict__ src,
                                const float* __restrict__ stats,
                                signed char* __restrict__ dst,
                                int Nelem, long long total) {
    long long idx = (long long)blockIdx.x * blockDim.x + threadIdx.x;
    if (idx >= total) return;
    int b = (int)(idx / Nelem);
    float mean = stats[b] / (float)Nelem;
    float v = src[idx] - mean;
    dst[idx] = (signed char)((v > 0.f) - (v < 0.f));
}

__global__ void maxpool2_kernel(const float* __restrict__ src,
                                float* __restrict__ dst,
                                int C, int H, int W, long long total) {
    long long idx = (long long)blockIdx.x * blockDim.x + threadIdx.x;
    if (idx >= total) return;
    int OW = W >> 1, OH = H >> 1;
    int ox = (int)(idx % OW);
    long long t = idx / OW;
    int oy = (int)(t % OH); t /= OH;
    int c = (int)(t % C);
    int b = (int)(t / C);
    const float* s = src + (((size_t)b * C + c) * H + oy * 2) * W + ox * 2;
    dst[idx] = fmaxf(fmaxf(s[0], s[1]), fmaxf(s[W], s[W + 1]));
}

__global__ void im2col_i8x4_kernel(const signed char* __restrict__ src,
                                   int* __restrict__ dst,
                                   int C, int H, int W, int KH, int KW,
                                   int OH, int OW, int Ppad, int K,
                                   int chunk_base, long long total_words) {
    long long idx = (long long)blockIdx.x * blockDim.x + threadIdx.x;
    if (idx >= total_words) return;
    const int Kw = K >> 2;
    int k0 = (int)(idx % Kw) << 2;
    long long t = idx / Kw;
    int p = (int)(t % Ppad);
    int b = (int)(t / Ppad);
    unsigned int res = 0;
    if (p < OH * OW) {
        int oy = p / OW, ox = p % OW;
        const signed char* sb =
            src + ((size_t)(chunk_base + b) * C) * H * W;
#pragma unroll
        for (int j = 0; j < 4; ++j) {
            int k = k0 + j;
            int c = k / (KH * KW);
            int r = k % (KH * KW);
            int ky = r / KW, kx = r % KW;
            unsigned char v =
                (unsigned char)sb[((size_t)c * H + oy + ky) * W + ox + kx];
            res |= (unsigned int)v << (8 * j);
        }
    }
    dst[idx] = (int)res;
}

__global__ void bin_gemm_wmma_kernel(const signed char* __restrict__ A,
                                     const signed char* __restrict__ W,
                                     float* __restrict__ out,
                                     const float* __restrict__ alpha,
                                     const float* __restrict__ bias,
                                     int K, int Mvalid, int N,
                                     int chunk_base, int out_mode, int relu,
                                     long long a_batch_stride) {
    const int lane = threadIdx.x & 31;
    const int wave = threadIdx.x >> 5;
    const int wm = wave & 3;
    const int wg = wave >> 2;
    const int m_tile = blockIdx.x * 4 + wm;
    const int n_group = blockIdx.y * 2 + wg;
    const int b = blockIdx.z;
    const int h = lane >> 4;
    const int l16 = lane & 15;

    const signed char* __restrict__ arow =
        A + (long long)b * a_batch_stride + (long long)(m_tile * 16 + l16) * K;
    const int o_base = n_group * 64 + l16;
    const signed char* __restrict__ brow0 = W + (long long)(o_base) * K;
    const signed char* __restrict__ brow1 = W + (long long)(o_base + 16) * K;
    const signed char* __restrict__ brow2 = W + (long long)(o_base + 32) * K;
    const signed char* __restrict__ brow3 = W + (long long)(o_base + 48) * K;

    v8i acc0 = {}, acc1 = {}, acc2 = {}, acc3 = {};
    for (int k0 = 0; k0 < K; k0 += 64) {
        __builtin_prefetch(arow + k0 + 256, 0, 1);
        const int2 a0 = *(const int2*)(arow + k0 + 16 * h);
        const int2 a1 = *(const int2*)(arow + k0 + 8 + 16 * h);
        const int2 a2 = *(const int2*)(arow + k0 + 32 + 16 * h);
        const int2 a3 = *(const int2*)(arow + k0 + 40 + 16 * h);
        v8i af = {a0.x, a0.y, a1.x, a1.y, a2.x, a2.y, a3.x, a3.y};
        {
            const int4 b0 = *(const int4*)(brow0 + k0 + 16 * h);
            const int4 b1 = *(const int4*)(brow0 + k0 + 32 + 16 * h);
            v8i bf = {b0.x, b0.y, b0.z, b0.w, b1.x, b1.y, b1.z, b1.w};
            acc0 = __builtin_amdgcn_wmma_i32_16x16x64_iu8(true, af, true, bf,
                                                          acc0, false, false);
        }
        {
            const int4 b0 = *(const int4*)(brow1 + k0 + 16 * h);
            const int4 b1 = *(const int4*)(brow1 + k0 + 32 + 16 * h);
            v8i bf = {b0.x, b0.y, b0.z, b0.w, b1.x, b1.y, b1.z, b1.w};
            acc1 = __builtin_amdgcn_wmma_i32_16x16x64_iu8(true, af, true, bf,
                                                          acc1, false, false);
        }
        {
            const int4 b0 = *(const int4*)(brow2 + k0 + 16 * h);
            const int4 b1 = *(const int4*)(brow2 + k0 + 32 + 16 * h);
            v8i bf = {b0.x, b0.y, b0.z, b0.w, b1.x, b1.y, b1.z, b1.w};
            acc2 = __builtin_amdgcn_wmma_i32_16x16x64_iu8(true, af, true, bf,
                                                          acc2, false, false);
        }
        {
            const int4 b0 = *(const int4*)(brow3 + k0 + 16 * h);
            const int4 b1 = *(const int4*)(brow3 + k0 + 32 + 16 * h);
            v8i bf = {b0.x, b0.y, b0.z, b0.w, b1.x, b1.y, b1.z, b1.w};
            acc3 = __builtin_amdgcn_wmma_i32_16x16x64_iu8(true, af, true, bf,
                                                          acc3, false, false);
        }
    }

    v8i accs[4] = {acc0, acc1, acc2, acc3};
#pragma unroll
    for (int j = 0; j < 4; ++j) {
        const int o = o_base + 16 * j;
        const float av = alpha[o];
        const float bv = bias ? bias[o] : 0.0f;
#pragma unroll
        for (int v = 0; v < 8; ++v) {
            int m = m_tile * 16 + v + 8 * h;
            if (m < Mvalid) {
                float val = ((float)accs[j][v] + bv) * av;
                if (relu) val = val > 0.f ? val : 0.f;
                if (out_mode == 0)
                    out[((size_t)(chunk_base + b) * N + o) * (size_t)Mvalid + m] =
                        val;
                else
                    out[(size_t)m * N + o] = val;
            }
        }
    }
}

extern "C" void kernel_launch(void* const* d_in, const int* in_sizes, int n_in,
                              void* d_out, int out_size, void* d_ws, size_t ws_size,
                              hipStream_t stream) {
    (void)in_sizes; (void)n_in; (void)out_size; (void)ws_size;
    const float* x        = (const float*)d_in[0];
    const float* conv1_w  = (const float*)d_in[1];
    const float* conv2_w  = (const float*)d_in[2];
    const float* conv2_al = (const float*)d_in[3];
    const float* conv3_w  = (const float*)d_in[4];
    const float* conv3_al = (const float*)d_in[5];
    const float* conv4_w  = (const float*)d_in[6];
    const float* conv4_al = (const float*)d_in[7];
    const float* fc1_w    = (const float*)d_in[8];
    const float* fc1_b    = (const float*)d_in[9];
    const float* fc1_al   = (const float*)d_in[10];
    float* out = (float*)d_out;

    const int B = 512;
    const int CHUNK = 64;

    char* ws = (char*)d_ws;
    size_t off = 0;
    auto take = [&](size_t bytes) -> char* {
        char* p = ws + off;
        off += (bytes + 255) & ~(size_t)255;
        return p;
    };
    signed char* w2   = (signed char*)take((size_t)128 * 3200);
    signed char* w3   = (signed char*)take((size_t)128 * 1152);
    signed char* w4   = (signed char*)take((size_t)128 * 1152);
    signed char* wfc  = (signed char*)take((size_t)512 * 2048);
    float*       h1   = (float*)take((size_t)B * 128 * 784 * 4);
    signed char* a2   = (signed char*)take((size_t)B * 128 * 784);
    signed char* col2 = (signed char*)take((size_t)CHUNK * 576 * 3200);
    float*       h2   = (float*)take((size_t)B * 128 * 576 * 4);
    float*       p2   = (float*)take((size_t)B * 128 * 144 * 4);
    signed char* a3   = (signed char*)take((size_t)B * 128 * 144);
    signed char* col3 = (signed char*)take((size_t)B * 128 * 1152);
    float*       h3   = (float*)take((size_t)B * 128 * 100 * 4);
    signed char* a4   = (signed char*)take((size_t)B * 128 * 100);
    signed char* col4 = (signed char*)take((size_t)B * 64 * 1152);
    float*       h4   = (float*)take((size_t)B * 128 * 64 * 4);
    float*       p4   = (float*)take((size_t)B * 2048 * 4);
    signed char* a5   = (signed char*)take((size_t)B * 2048);
    float*       part = (float*)take((size_t)B * 64 * 4);
    float*       stat = (float*)take((size_t)B * 4);

    pack_sign_i8_kernel<<<(128 * 3200 + 255) / 256, 256, 0, stream>>>(conv2_w, w2, 128 * 3200);
    pack_sign_i8_kernel<<<(128 * 1152 + 255) / 256, 256, 0, stream>>>(conv3_w, w3, 128 * 1152);
    pack_sign_i8_kernel<<<(128 * 1152 + 255) / 256, 256, 0, stream>>>(conv4_w, w4, 128 * 1152);
    pack_sign_i8_kernel<<<(512 * 2048 + 255) / 256, 256, 0, stream>>>(fc1_w, wfc, 512 * 2048);

    {
        int total = B * 128 * 784;
        conv1_relu_kernel<<<(total + 255) / 256, 256, 0, stream>>>(x, conv1_w, h1, total);
    }
    reduce_sum_partial_kernel<<<dim3(64, B), 256, 0, stream>>>(h1, part, 128 * 784);
    reduce_sum_final_kernel<<<B, 256, 0, stream>>>(part, stat, 64);
    {
        long long total = (long long)B * 128 * 784;
        binarize_kernel<<<(int)((total + 255) / 256), 256, 0, stream>>>(h1, stat, a2, 128 * 784, total);
    }
    for (int cb = 0; cb < B; cb += CHUNK) {
        long long tw = (long long)CHUNK * 576 * 800;
        im2col_i8x4_kernel<<<(int)((tw + 255) / 256), 256, 0, stream>>>(
            a2, (int*)col2, 128, 28, 28, 5, 5, 24, 24, 576, 3200, cb, tw);
        bin_gemm_wmma_kernel<<<dim3(9, 1, CHUNK), 256, 0, stream>>>(
            col2, w2, h2, conv2_al, nullptr, 3200, 576, 128, cb, 0, 1,
            (long long)576 * 3200);
    }
    {
        long long total = (long long)B * 128 * 144;
        maxpool2_kernel<<<(int)((total + 255) / 256), 256, 0, stream>>>(h2, p2, 128, 24, 24, total);
    }
    reduce_sum_partial_kernel<<<dim3(64, B), 256, 0, stream>>>(p2, part, 128 * 144);
    reduce_sum_final_kernel<<<B, 256, 0, stream>>>(part, stat, 64);
    {
        long long total = (long long)B * 128 * 144;
        binarize_kernel<<<(int)((total + 255) / 256), 256, 0, stream>>>(p2, stat, a3, 128 * 144, total);
    }
    {
        long long tw = (long long)B * 128 * 288;
        im2col_i8x4_kernel<<<(int)((tw + 255) / 256), 256, 0, stream>>>(
            a3, (int*)col3, 128, 12, 12, 3, 3, 10, 10, 128, 1152, 0, tw);
        bin_gemm_wmma_kernel<<<dim3(2, 1, B), 256, 0, stream>>>(
            col3, w3, h3, conv3_al, nullptr, 1152, 100, 128, 0, 0, 1,
            (long long)128 * 1152);
    }
    reduce_sum_partial_kernel<<<dim3(64, B), 256, 0, stream>>>(h3, part, 128 * 100);
    reduce_sum_final_kernel<<<B, 256, 0, stream>>>(part, stat, 64);
    {
        long long total = (long long)B * 128 * 100;
        binarize_kernel<<<(int)((total + 255) / 256), 256, 0, stream>>>(h3, stat, a4, 128 * 100, total);
    }
    {
        long long tw = (long long)B * 64 * 288;
        im2col_i8x4_kernel<<<(int)((tw + 255) / 256), 256, 0, stream>>>(
            a4, (int*)col4, 128, 10, 10, 3, 3, 8, 8, 64, 1152, 0, tw);
        bin_gemm_wmma_kernel<<<dim3(1, 1, B), 256, 0, stream>>>(
            col4, w4, h4, conv4_al, nullptr, 1152, 64, 128, 0, 0, 1,
            (long long)64 * 1152);
    }
    {
        long long total = (long long)B * 128 * 16;
        maxpool2_kernel<<<(int)((total + 255) / 256), 256, 0, stream>>>(h4, p4, 128, 8, 8, total);
    }
    reduce_sum_partial_kernel<<<dim3(8, B), 256, 0, stream>>>(p4, part, 2048);
    reduce_sum_final_kernel<<<B, 256, 0, stream>>>(part, stat, 8);
    {
        long long total = (long long)B * 2048;
        binarize_kernel<<<(int)((total + 255) / 256), 256, 0, stream>>>(p4, stat, a5, 2048, total);
    }
    bin_gemm_wmma_kernel<<<dim3(8, 4, 1), 256, 0, stream>>>(
        a5, wfc, out, fc1_al, fc1_b, 2048, 512, 512, 0, 1, 0, 0);
}
